// TemporalEncoder_1975684956295
// MI455X (gfx1250) — compile-verified, matrix operand order checked by reading, not run on hardware
//
#include <hip/hip_runtime.h>
#include <hip/hip_bf16.h>

typedef __attribute__((ext_vector_type(16))) _Float16 v16h;
typedef __attribute__((ext_vector_type(8)))  _Float16 v8h;
typedef __attribute__((ext_vector_type(8)))  float    v8f;
typedef __attribute__((ext_vector_type(4)))  int      v4i;

#define BB      16
#define LL      2048
#define DM      128
#define DI      256
#define DSTATE  16
#define NROW    (BB*LL)
#define NCHUNK  64
#define CLEN    32

// ---- gfx1250 async global->LDS path (guarded; falls back to b128 copies) ----
#if defined(__AMDGCN__) && __has_builtin(__builtin_amdgcn_global_load_async_to_lds_b128) && __has_builtin(__builtin_amdgcn_s_wait_asynccnt)
#define USE_ASYNC_LDS 1
#else
#define USE_ASYNC_LDS 0
#endif

#if USE_ASYNC_LDS
typedef __attribute__((address_space(1))) v4i gas_v4i;   // global int4
typedef __attribute__((address_space(3))) v4i las_v4i;   // LDS int4
__device__ __forceinline__ void async_cp16(const void* g, void* l) {
  __builtin_amdgcn_global_load_async_to_lds_b128(
      (gas_v4i*)(void*)g, (las_v4i*)(void*)l, 0, 0);
}
#endif

// copy `n16` 16-byte units from global to LDS with 256 threads (n16 % 256 == 0)
__device__ __forceinline__ void tile_copy(const void* g, void* l, int n16, int tid) {
#if USE_ASYNC_LDS
  const uint4* gp = (const uint4*)g;
  uint4* lp = (uint4*)l;
  for (int i = tid; i < n16; i += 256) async_cp16(gp + i, lp + i);
  __builtin_amdgcn_s_wait_asynccnt(0);
#else
  const uint4* gp = (const uint4*)g;
  uint4* lp = (uint4*)l;
  for (int i = tid; i < n16; i += 256) lp[i] = gp[i];
#endif
}

__device__ __forceinline__ v16h frag_from2(v8h lo, v8h hi) {
  return __builtin_shufflevector(lo, hi, 0,1,2,3,4,5,6,7,8,9,10,11,12,13,14,15);
}

// ---------------- K0: pre-swizzle w2 / in_w into fragment-ready f16 ----------------
// layout per 64-col chunk: [ct(4)][kb(4)][lane(32)][j(16)]  (8192 halves = 16KB)
__global__ __launch_bounds__(256) void k0_prep(
    const float* __restrict__ w2, const float* __restrict__ in_w,
    _Float16* __restrict__ w2f16, _Float16* __restrict__ inwf16)
{
  int idx = blockIdx.x * 256 + threadIdx.x;          // 81920 total
  const float* W; _Float16* O; int ldw; int chunk; int f;
  if (idx < 2 * 8192) { W = w2;   O = w2f16;  ldw = 128; chunk = idx >> 13; f = idx & 8191; }
  else { int t2 = idx - 16384; W = in_w; O = inwf16; ldw = 512; chunk = t2 >> 13; f = t2 & 8191; }
  int j = f & 15, lane = (f >> 4) & 31, kb = (f >> 9) & 3, ct = (f >> 11) & 3;
  int nsel = lane & 15, khalf = lane >> 4;
  int k   = kb * 32 + (j >> 3) * 16 + khalf * 8 + (j & 7);
  int col = chunk * 64 + ct * 16 + nsel;
  O[chunk * 8192 + f] = (_Float16)W[k * ldw + col];
}

// ---------------- K1: calendar + MLP -> h (f32 + f16 copies) ----------------
__global__ __launch_bounds__(256) void k1_embed(
    const float* __restrict__ ts, const float* __restrict__ seq,
    const float* __restrict__ w1, const float* __restrict__ b1,
    const _Float16* __restrict__ w2f16, const float* __restrict__ b2,
    float* __restrict__ h_out, _Float16* __restrict__ h16)
{
  __shared__ float s_cal[128][12];
  __shared__ __align__(16) _Float16 s_a1[128][128];
  __shared__ __align__(16) _Float16 s_wf[4][4][32][16];
  const int tid = threadIdx.x;
  const int rowbase = blockIdx.x * 128;

  if (tid < 128) {
    const float t = ts[rowbase + tid];
    const float TWO_PI = 6.28318530717958647692f;
    float normalized = (t - 1577836800.0f) * (1.0f / (365.25f * 86400.0f * 10.0f));
    float hour  = fmodf(t, 86400.0f) * (1.0f / 86400.0f);
    float days  = t * (1.0f / 86400.0f);
    float dayf  = fmodf(days, 365.25f) * (1.0f / 365.25f);
    float monf  = fmodf(days, 30.44f)  * (1.0f / 30.44f);
    s_cal[tid][0] = __sinf(TWO_PI * hour);
    s_cal[tid][1] = __cosf(TWO_PI * hour);
    s_cal[tid][2] = __sinf(TWO_PI * dayf);
    s_cal[tid][3] = __cosf(TWO_PI * dayf);
    s_cal[tid][4] = __sinf(TWO_PI * monf);
    s_cal[tid][5] = __cosf(TWO_PI * monf);
    s_cal[tid][6] = __sinf(TWO_PI * dayf * 4.0f);
    s_cal[tid][7] = __cosf(TWO_PI * dayf * 4.0f);
    s_cal[tid][8] = normalized;
  }
  __syncthreads();

  for (int i = 0; i < 64; ++i) {                       // layer1 + SiLU -> f16 LDS
    int o = i * 256 + tid;
    int r = o >> 7, c = o & 127;
    float acc = b1[c];
#pragma unroll
    for (int k = 0; k < 9; ++k) acc += s_cal[r][k] * w1[k * 128 + c];
    float sv = acc / (1.0f + __expf(-acc));
    s_a1[r][c] = (_Float16)sv;
  }
  __syncthreads();

  const int lane  = tid & 31;
  const int wv    = tid >> 5;
  const int nsel  = lane & 15;
  const int khalf = lane >> 4;

  v16h afr[4];                                         // A frags: 2x ds_load_b128 each
#pragma unroll
  for (int kb = 0; kb < 4; ++kb) {
    const v8h* pa = (const v8h*)&s_a1[wv * 16 + nsel][kb * 32 + khalf * 8];
    afr[kb] = frag_from2(pa[0], pa[2]);
  }

  for (int chunk = 0; chunk < 2; ++chunk) {
    __syncthreads();
    tile_copy(w2f16 + (size_t)chunk * 8192, &s_wf[0][0][0][0], 1024, tid);
    __syncthreads();
    for (int ct = 0; ct < 4; ++ct) {
      v8f acc = {};
#pragma unroll
      for (int kb = 0; kb < 4; ++kb) {
        const v8h* pb = (const v8h*)&s_wf[ct][kb][lane][0];
        v16h bfr = frag_from2(pb[0], pb[1]);
        acc = __builtin_amdgcn_wmma_f32_16x16x32_f16(false, afr[kb], false, bfr,
                                                     (short)0, acc, false, false);
      }
      int col = chunk * 64 + ct * 16 + nsel;
#pragma unroll
      for (int i = 0; i < 8; ++i) {
        int row = rowbase + wv * 16 + khalf * 8 + i;
        float v = acc[i] + b2[col] + seq[(size_t)row * 128 + col];
        h_out[(size_t)row * 128 + col] = v;
        h16[(size_t)row * 128 + col]   = (_Float16)v;
      }
    }
  }
}

// ---------------- K2: xz = h @ in_w ----------------
__global__ __launch_bounds__(256) void k2_xz(
    const _Float16* __restrict__ h16, const _Float16* __restrict__ inwf16,
    float* __restrict__ x_raw, float* __restrict__ z_last)
{
  __shared__ __align__(16) _Float16 s_h[128][128];
  __shared__ __align__(16) _Float16 s_wf[4][4][32][16];
  const int tid = threadIdx.x;
  const int rowbase = blockIdx.x * 128;

  tile_copy(h16 + (size_t)rowbase * 128, &s_h[0][0], 2048, tid);  // 32KB tile
  __syncthreads();

  const int lane  = tid & 31;
  const int wv    = tid >> 5;
  const int nsel  = lane & 15;
  const int khalf = lane >> 4;

  v16h afr[4];
#pragma unroll
  for (int kb = 0; kb < 4; ++kb) {
    const v8h* pa = (const v8h*)&s_h[wv * 16 + nsel][kb * 32 + khalf * 8];
    afr[kb] = frag_from2(pa[0], pa[2]);
  }

  const int nchunks = ((blockIdx.x & 15) == 15) ? 8 : 4;  // z half only at t==L-1
  for (int chunk = 0; chunk < nchunks; ++chunk) {
    __syncthreads();
    tile_copy(inwf16 + (size_t)chunk * 8192, &s_wf[0][0][0][0], 1024, tid);
    __syncthreads();
    for (int ct = 0; ct < 4; ++ct) {
      v8f acc = {};
#pragma unroll
      for (int kb = 0; kb < 4; ++kb) {
        const v8h* pb = (const v8h*)&s_wf[ct][kb][lane][0];
        v16h bfr = frag_from2(pb[0], pb[1]);
        acc = __builtin_amdgcn_wmma_f32_16x16x32_f16(false, afr[kb], false, bfr,
                                                     (short)0, acc, false, false);
      }
      int colg = chunk * 64 + ct * 16 + nsel;
#pragma unroll
      for (int i = 0; i < 8; ++i) {
        int row = rowbase + wv * 16 + khalf * 8 + i;
        float v = acc[i];
        if (colg < 256) {
          x_raw[(size_t)row * 256 + colg] = v;
        } else {
          if ((row & (LL - 1)) == LL - 1)
            z_last[(row >> 11) * 256 + (colg - 256)] = v;
        }
      }
    }
  }
}

// ---------------- K3: causal depthwise conv + SiLU ----------------
__global__ __launch_bounds__(256) void k3_conv(
    const float* __restrict__ x_raw, const float* __restrict__ conv_w,
    const float* __restrict__ conv_b, float* __restrict__ x_act)
{
  int idx = blockIdx.x * 256 + threadIdx.x;
  int c   = idx & 255;
  int row = idx >> 8;
  int t   = row & (LL - 1);
  float acc = conv_b[c];
#pragma unroll
  for (int k = 0; k < 4; ++k) {
    int tt = t - 3 + k;
    if (tt >= 0) acc += x_raw[(size_t)(row - 3 + k) * 256 + c] * conv_w[c * 4 + k];
  }
  x_act[idx] = acc / (1.0f + __expf(-acc));
}

// ---------------- K4: x_dbl ; dt softplus ; B ; C(last) ----------------
__global__ __launch_bounds__(256) void k4_proj(
    const float* __restrict__ x_act, const float* __restrict__ xproj_w,
    const float* __restrict__ dt_w, const float* __restrict__ dt_b,
    float* __restrict__ dt_sp, float* __restrict__ Bc,
    float* __restrict__ Cc_last)
{
  __shared__ float s_x[32][257];
  __shared__ float s_d[32][40];
  const int tid = threadIdx.x;
  const int rowbase = blockIdx.x * 32;

  for (int i = 0; i < 32; ++i) {
    int o = i * 256 + tid;
    int r = o >> 8, c = o & 255;
    s_x[r][c] = x_act[(size_t)(rowbase + r) * 256 + c];
  }
  __syncthreads();
  for (int i = 0; i < 5; ++i) {
    int o = i * 256 + tid;
    int r = o / 40, c = o % 40;
    float acc = 0.0f;
    for (int k = 0; k < 256; ++k) acc += s_x[r][k] * xproj_w[k * 40 + c];
    s_d[r][c] = acc;
  }
  __syncthreads();
  for (int i = 0; i < 32; ++i) {
    int o = i * 256 + tid;
    int r = o >> 8, c = o & 255;
    float acc = dt_b[c];
#pragma unroll
    for (int k = 0; k < 8; ++k) acc += s_d[r][k] * dt_w[k * 256 + c];
    float sp = (acc > 20.0f) ? acc : log1pf(__expf(acc));
    dt_sp[(size_t)(rowbase + r) * 256 + c] = sp;
  }
  for (int i = 0; i < 2; ++i) {
    int o = i * 256 + tid;
    int r = o >> 4, n = o & 15;
    int row = rowbase + r;
    Bc[(size_t)row * 16 + n] = s_d[r][8 + n];
    if ((row & (LL - 1)) == LL - 1)
      Cc_last[(row >> 11) * 16 + n] = s_d[r][24 + n];
  }
}

// ---------------- K5: chunked parallel scan; block=(b,chunk,n), thread=e ----------------
__global__ __launch_bounds__(256) void k5_chunkscan(
    const float* __restrict__ dt_sp, const float* __restrict__ x_act,
    const float* __restrict__ Bc, const float* __restrict__ A_log,
    float* __restrict__ chP, float* __restrict__ chH)
{
  const int blk = blockIdx.x;                // BB * NCHUNK * DSTATE
  const int n   = blk & 15;
  const int c   = (blk >> 4) & 63;
  const int b   = blk >> 10;
  const int e   = threadIdx.x;

  const float A = -__expf(A_log[e * 16 + n]);
  float P = 1.0f, H = 0.0f;
  const int base = b * LL + c * CLEN;
  __builtin_prefetch(&dt_sp[(size_t)base * 256 + e], 0, 0);
  __builtin_prefetch(&x_act[(size_t)base * 256 + e], 0, 0);
#pragma unroll 4
  for (int s = 0; s < CLEN; ++s) {
    const size_t row = (size_t)(base + s);
    float dt = dt_sp[row * 256 + e];      // coalesced
    float u  = x_act[row * 256 + e];      // coalesced
    float Bv = Bc[row * 16 + n];          // uniform broadcast
    float a  = __expf(dt * A);
    H = a * H + (dt * u) * Bv;
    P *= a;
  }
  size_t o = ((size_t)(b * NCHUNK + c) * DSTATE + n) * DI + e;   // [b][c][n][e]
  chP[o] = P;
  chH[o] = H;
}

// ---------------- K6: combine + readout + head + LayerNorm ----------------
__global__ __launch_bounds__(256) void k6_final(
    const float* __restrict__ chP, const float* __restrict__ chH,
    const float* __restrict__ Cc_last, const float* __restrict__ z_last,
    const float* __restrict__ x_act, const float* __restrict__ Dvec,
    const float* __restrict__ ssm_out_w, const float* __restrict__ h_f32,
    const float* __restrict__ out_w, const float* __restrict__ out_b,
    const float* __restrict__ ln_g, const float* __restrict__ ln_b,
    float* __restrict__ out)
{
  __shared__ float s_y[256];
  __shared__ float s_h[128];
  __shared__ float s_red[256];
  const int b = blockIdx.x;
  const int e = threadIdx.x;
  const int lastrow = b * LL + (LL - 1);

  float y = 0.0f;
  for (int n = 0; n < DSTATE; ++n) {
    float H = 0.0f;
    for (int c = 0; c < NCHUNK; ++c) {
      size_t o = ((size_t)(b * NCHUNK + c) * DSTATE + n) * DI + e;  // coalesced in e
      H = chP[o] * H + chH[o];
    }
    y += H * Cc_last[b * 16 + n];
  }
  y += x_act[(size_t)lastrow * 256 + e] * Dvec[e];
  float z = z_last[b * 256 + e];
  y *= z / (1.0f + __expf(-z));
  s_y[e] = y;
  __syncthreads();

  if (e < 128) {
    float acc = 0.0f;
    for (int k = 0; k < 256; ++k) acc += s_y[k] * ssm_out_w[k * 128 + e];
    s_h[e] = h_f32[(size_t)lastrow * 128 + e] + acc;
  }
  __syncthreads();

  float ov = out_b[e];
  for (int k = 0; k < 128; ++k) ov += s_h[k] * out_w[k * 256 + e];

  s_red[e] = ov;
  __syncthreads();
  for (int st = 128; st > 0; st >>= 1) {
    if (e < st) s_red[e] += s_red[e + st];
    __syncthreads();
  }
  float mu = s_red[0] * (1.0f / 256.0f);
  __syncthreads();
  float dd = ov - mu;
  s_red[e] = dd * dd;
  __syncthreads();
  for (int st = 128; st > 0; st >>= 1) {
    if (e < st) s_red[e] += s_red[e + st];
    __syncthreads();
  }
  float var = s_red[0] * (1.0f / 256.0f);
  out[b * 256 + e] = dd * rsqrtf(var + 1e-5f) * ln_g[e] + ln_b[e];
}

// ---------------- launch ----------------
extern "C" void kernel_launch(void* const* d_in, const int* in_sizes, int n_in,
                              void* d_out, int out_size, void* d_ws, size_t ws_size,
                              hipStream_t stream)
{
  (void)in_sizes; (void)n_in; (void)out_size; (void)ws_size;
  const float* ts       = (const float*)d_in[0];
  const float* seq      = (const float*)d_in[1];
  const float* w1       = (const float*)d_in[2];
  const float* b1       = (const float*)d_in[3];
  const float* w2       = (const float*)d_in[4];
  const float* b2       = (const float*)d_in[5];
  const float* in_w     = (const float*)d_in[6];
  const float* conv_w   = (const float*)d_in[7];
  const float* conv_b   = (const float*)d_in[8];
  const float* xproj_w  = (const float*)d_in[9];
  const float* dt_w     = (const float*)d_in[10];
  const float* dt_b     = (const float*)d_in[11];
  const float* A_log    = (const float*)d_in[12];
  const float* Dvec     = (const float*)d_in[13];
  const float* ssm_out_w= (const float*)d_in[14];
  const float* out_w    = (const float*)d_in[15];
  const float* out_b    = (const float*)d_in[16];
  const float* ln_g     = (const float*)d_in[17];
  const float* ln_b     = (const float*)d_in[18];

  float* ws = (float*)d_ws;
  float* h_f32   = ws;                                        // NROW*128
  float* x_raw   = h_f32   + (size_t)NROW * DM;               // NROW*256
  float* x_act   = x_raw   + (size_t)NROW * DI;               // NROW*256
  float* dt_sp   = x_act   + (size_t)NROW * DI;               // NROW*256
  float* Bc      = dt_sp   + (size_t)NROW * DI;               // NROW*16
  float* Cc_last = Bc      + (size_t)NROW * DSTATE;           // B*16
  float* z_last  = Cc_last + (size_t)BB * DSTATE;             // B*256
  float* chP     = z_last  + (size_t)BB * DI;                 // B*64*16*256
  float* chH     = chP     + (size_t)BB * NCHUNK * DI * DSTATE;
  _Float16* h16    = (_Float16*)(chH + (size_t)BB * NCHUNK * DI * DSTATE); // NROW*128 halves
  _Float16* w2f16  = h16 + (size_t)NROW * DM;                 // 2*8192 halves
  _Float16* inwf16 = w2f16 + 2 * 8192;                        // 8*8192 halves

  k0_prep    <<<320,        256, 0, stream>>>(w2, in_w, w2f16, inwf16);
  k1_embed   <<<NROW / 128, 256, 0, stream>>>(ts, seq, w1, b1, w2f16, b2, h_f32, h16);
  k2_xz      <<<NROW / 128, 256, 0, stream>>>(h16, inwf16, x_raw, z_last);
  k3_conv    <<<NROW,       256, 0, stream>>>(x_raw, conv_w, conv_b, x_act);
  k4_proj    <<<NROW / 32,  256, 0, stream>>>(x_act, xproj_w, dt_w, dt_b, dt_sp, Bc, Cc_last);
  k5_chunkscan<<<BB * NCHUNK * DSTATE, 256, 0, stream>>>(dt_sp, x_act, Bc, A_log, chP, chH);
  k6_final   <<<BB, 256, 0, stream>>>(chP, chH, Cc_last, z_last, x_act, Dvec,
                                      ssm_out_w, h_f32, out_w, out_b, ln_g, ln_b,
                                      (float*)d_out);
}
